// Conttime_18528488915448
// MI455X (gfx1250) — compile-verified, matrix operand order checked by reading, not run on hardware
//
#include <hip/hip_runtime.h>
#include <hip/hip_bf16.h>

// ---------------------------------------------------------------------------
// Continuous-time LSTM scan for MI455X (gfx1250, wave32, WMMA).
//
// Sizes: B=128, S=1024 (T=1023 steps), H=256, 7H=1792, N_TYPES+1=65.
//
// Key transformations:
//  * Fold emb_W @ W[:H,:] + bias into a 65x1792 table (input term is a lookup).
//  * Sequential GEMM per step is h[128,256] @ Wh[256,1792] in bf16 WMMA
//    (v_wmma_f32_16x16x32_bf16, f32 accum), Wh pre-packed in fragment order.
//  * 128 persistent blocks = 8 batch-groups x 16 hid-col-groups. Each block
//    owns c / c_bar for its 16x16 slab in registers for all 1023 steps; only
//    h crosses blocks (double-buffered global, L2-resident, 64KB bf16).
//  * Batch groups are fully independent -> 8 independent 16-block barriers
//    (per-group atomic counter in its own cacheline, memset to 0 per call).
//  * Latency hiding on the sequential critical path: types/dtime prefetched
//    one step ahead; tbl row loads issued before the WMMA chain; all 8 A
//    fragments loaded as one LDS clause, enforced by sched_barrier, so the
//    8 WMMAs chain back-to-back on the accumulator.
// ---------------------------------------------------------------------------

typedef __attribute__((ext_vector_type(16))) __bf16 v16bf;
typedef __attribute__((ext_vector_type(8)))  __bf16 v8bf;
typedef __attribute__((ext_vector_type(8)))  float  v8f;

#define HID   256
#define SEVH  1792
#define BATCH 128
#define SEQ   1024
#define TSTEP 1023
#define NTY   65
#define NBLK  128

__device__ __forceinline__ unsigned short f2bf(float x) {
    unsigned u = __float_as_uint(x);
    unsigned r = (u + 0x7FFFu + ((u >> 16) & 1u)) >> 16;   // RNE
    return (unsigned short)r;
}

__device__ __forceinline__ float sigf(float x) {
    return 1.0f / (1.0f + __expf(-x));
}

// ---------------------------------------------------------------------------
// Prep 1: input table  tbl[ty][col] = emb_W[ty,:] @ W[0:256, col] + bias[col]
// ---------------------------------------------------------------------------
__global__ void prep_tbl(const float* __restrict__ embW,
                         const float* __restrict__ W,
                         const float* __restrict__ bias,
                         float* __restrict__ tbl) {
    int id = blockIdx.x * blockDim.x + threadIdx.x;
    if (id >= NTY * SEVH) return;
    int ty = id / SEVH, col = id % SEVH;
    float acc = bias[col];
    const float* e = embW + ty * HID;
    for (int k = 0; k < HID; ++k)
        acc = fmaf(e[k], W[k * SEVH + col], acc);
    tbl[id] = acc;
}

// ---------------------------------------------------------------------------
// Prep 2: pack recurrent weights W[256:512, :] into bf16 WMMA B-fragment order.
// Layout: [jg(16)][gate(7)][kc(8)][lane(32)][elem(16)]  (elem = bf16)
// B frag (32x16, K x N): lane<16 -> col=lane,    e0..7=K(kb..kb+7),  e8..15=K(kb+16..kb+23)
//                        lane>=16-> col=lane-16, e0..7=K(kb+8..+15), e8..15=K(kb+24..+31)
// ---------------------------------------------------------------------------
__global__ void prep_wbf(const float* __restrict__ W,
                         unsigned short* __restrict__ wbf) {
    int id = blockIdx.x * blockDim.x + threadIdx.x;   // < 16*7*8*32*16 = 458752
    int e    = id & 15;
    int lane = (id >> 4) & 31;
    int kc   = (id >> 9) & 7;
    int g7   = id >> 12;            // 0..111
    int gate = g7 % 7;
    int jg   = g7 / 7;
    int col  = gate * HID + jg * 16 + (lane & 15);
    int ko = e & 7, half = e >> 3;
    int k = kc * 32 + ((lane < 16) ? (half ? 16 + ko : ko)
                                   : (half ? 24 + ko : 8 + ko));
    wbf[id] = f2bf(W[(HID + k) * SEVH + col]);
}

// ---------------------------------------------------------------------------
// Persistent scan kernel. grid = 128 blocks (8 independent groups of 16,
// each group must be co-resident). block = 256 threads = 8 waves.
// Waves 0..6: one 16x16 WMMA gate tile each.
// ---------------------------------------------------------------------------
__global__ __launch_bounds__(256, 1)
void ctlstm_scan(const int*   __restrict__ types,
                 const float* __restrict__ dtime,
                 const unsigned short* __restrict__ wbf,
                 const float* __restrict__ tbl,
                 unsigned short* __restrict__ h0,
                 unsigned short* __restrict__ h1,
                 unsigned int*  __restrict__ syncc,
                 float* __restrict__ out) {
    __shared__ __align__(16) unsigned short a_lds[16 * HID];   // 8 KB: h slab bf16
    __shared__ __align__(16) float          g_lds[7 * 16 * 16];// 7 KB: gate tiles f32

    const int tid  = threadIdx.x;
    const int w    = tid >> 5;          // wave id 0..7
    const int lane = tid & 31;
    const int mg   = blockIdx.x >> 4;   // batch group 0..7
    const int jg   = blockIdx.x & 15;   // hidden-col group 0..15

    unsigned int* mycnt = syncc + (size_t)mg * 64;   // per-group counter, own line

    // Preload B fragments (resident in VGPRs for the whole scan): 8 K-chunks.
    v16bf bfr[8];
    if (w < 7) {
        const v16bf* src = (const v16bf*)wbf + ((size_t)(jg * 7 + w) * 8) * 32 + lane;
#pragma unroll
        for (int kc = 0; kc < 8; ++kc) bfr[kc] = src[kc * 32];
    }

    // Per-thread recurrent cell state for element (batch=b_el, hid).
    const int r_el = tid >> 4, n_el = tid & 15;
    const int b_el = mg * 16 + r_el;
    const int hid  = jg * 16 + n_el;
    float c = 0.0f, cb = 0.0f;

    const size_t KS = (size_t)TSTEP * BATCH * HID;   // per-output-channel stride

    // Prefetch step-0 event data (independent of any cross-block state).
    int   ty = types[(size_t)b_el * SEQ];
    float dt = dtime[(size_t)b_el * SEQ + 1];

    for (int t = 0; t < TSTEP; ++t) {
        const unsigned short* hsrc = (t & 1) ? h1 : h0;
        unsigned short*       hdst = (t & 1) ? h0 : h1;

        // Stage h rows [mg*16, mg*16+16), all 256 cols, into LDS (bf16, 8KB).
        {
            int row = tid >> 4, cg = tid & 15;
            const uint4* gp = (const uint4*)(hsrc + (size_t)(mg * 16 + row) * HID + cg * 16);
            uint4 u0 = gp[0];
            uint4 u1 = gp[1];
            uint4* lp = (uint4*)(a_lds + row * HID + cg * 16);
            lp[0] = u0;
            lp[1] = u1;
        }

        // Issue table-row loads now (ty was prefetched last step) so they
        // fly during the WMMA chain below.
        const float* tb = tbl + (size_t)ty * SEVH + hid;
        float tb0 = tb[0];
        float tb1 = tb[256];
        float tb2 = tb[512];
        float tb3 = tb[768];
        float tb4 = tb[1024];
        float tb5 = tb[1280];
        float tb6 = tb[1536];

        __syncthreads();

        // GEMM: each of waves 0..6 computes one 16x16 tile of g = h @ Wh.
        // All 16 ds_load_b128 issue as one clause into distinct VGPRs; the
        // sched_barrier stops the scheduler from re-sinking them between the
        // WMMAs, so the 8 WMMAs chain back-to-back on the accumulator.
        if (w < 7) {
            const int ar   = lane & 15;
            const int kofs = (lane < 16) ? 0 : 8;
            union { v16bf v; v8bf h[2]; } au[8];
#pragma unroll
            for (int kc = 0; kc < 8; ++kc) {
                const unsigned short* ap = a_lds + ar * HID + kc * 32 + kofs;
                au[kc].h[0] = *(const v8bf*)(ap);
                au[kc].h[1] = *(const v8bf*)(ap + 16);
            }
            __builtin_amdgcn_sched_barrier(0);   // keep loads above, WMMAs below
            v8f acc = {};
#pragma unroll
            for (int kc = 0; kc < 8; ++kc)
                acc = __builtin_amdgcn_wmma_f32_16x16x32_bf16(
                        false, au[kc].v, false, bfr[kc], (short)0, acc, false, false);

            const int nn = lane & 15;
            const int mo = (lane < 16) ? 0 : 8;
#pragma unroll
            for (int r8 = 0; r8 < 8; ++r8)
                g_lds[w * 256 + (mo + r8) * 16 + nn] = acc[r8];
        }
        __syncthreads();

        // Elementwise recurrence: thread owns element (b_el, hid).
        {
            const float* gl = g_lds + r_el * 16 + n_el;

            float g0 = gl[0]    + tb0;
            float g1 = gl[256]  + tb1;
            float g2 = gl[512]  + tb2;
            float g3 = gl[768]  + tb3;
            float g4 = gl[1024] + tb4;
            float g5 = gl[1280] + tb5;
            float g6 = gl[1536] + tb6;

            float gi  = sigf(g0);
            float gf  = sigf(g1);
            float go  = sigf(g2);
            float gz  = tanhf(g3);
            float gib = sigf(g4);
            float gfb = sigf(g5);
            float zz  = 0.1f * g6;
            float gd  = (zz > 15.0f) ? g6 : 10.0f * log1pf(__expf(zz));

            float ci   = gf * c + gi * gz;        // cell right after event
            float cbi  = gfb * cb + gib * gz;     // target cell
            float cdec = cbi + (ci - cbi) * __expf(-gd * dt);
            float hdec = go * tanhf(cdec);

            c  = cdec;                            // carry = c_dec
            cb = cbi;

            size_t ob = ((size_t)t * BATCH + b_el) * HID + hid;
            out[ob]          = hdec;
            out[ob + KS]     = ci;
            out[ob + 2 * KS] = cbi;
            out[ob + 3 * KS] = gd;
            out[ob + 4 * KS] = go;

            hdst[(size_t)b_el * HID + hid] = f2bf(hdec);
        }

        // Prefetch next step's event data before the barrier (hides latency).
        if (t + 1 < TSTEP) {
            ty = types[(size_t)b_el * SEQ + t + 1];
            dt = dtime[(size_t)b_el * SEQ + t + 2];
        }

        // Per-batch-group barrier (16 blocks, monotonic counter).
        __threadfence();
        __syncthreads();
        if (tid == 0) {
            atomicAdd(mycnt, 1u);
            unsigned target = 16u * (unsigned)(t + 1);
            while (__hip_atomic_load(mycnt, __ATOMIC_RELAXED,
                                     __HIP_MEMORY_SCOPE_AGENT) < target) {
                __builtin_amdgcn_s_sleep(1);
            }
        }
        __syncthreads();
        __threadfence();
    }
}

// ---------------------------------------------------------------------------
// Workspace layout (bytes):
//   [0,      2048)     8 sync counters (256B apart, one per batch group)
//   [2048,   67584)    h buffer 0 (128*256 bf16)
//   [67584,  133120)   h buffer 1
//   [133120, 1050624)  packed bf16 weights (458752 * 2)
//   [1050624,1516544)  input table (65*1792 f32)
// ---------------------------------------------------------------------------
extern "C" void kernel_launch(void* const* d_in, const int* in_sizes, int n_in,
                              void* d_out, int out_size, void* d_ws, size_t ws_size,
                              hipStream_t stream) {
    const int*   types = (const int*)d_in[0];
    const float* dtime = (const float*)d_in[1];
    const float* embW  = (const float*)d_in[2];
    const float* W     = (const float*)d_in[3];
    const float* bias  = (const float*)d_in[4];

    char* ws = (char*)d_ws;
    unsigned int*   syncc = (unsigned int*)(ws + 0);
    unsigned short* h0    = (unsigned short*)(ws + 2048);
    unsigned short* h1    = (unsigned short*)(ws + 67584);
    unsigned short* wbf   = (unsigned short*)(ws + 133120);
    float*          tbl   = (float*)(ws + 1050624);

    hipMemsetAsync(syncc, 0, 2048, stream);                      // reset barriers
    hipMemsetAsync(h0, 0, (size_t)BATCH * HID * 2, stream);      // h init = 0

    prep_tbl<<<(NTY * SEVH + 255) / 256, 256, 0, stream>>>(embW, W, bias, tbl);
    prep_wbf<<<(16 * 7 * 8 * 32 * 16) / 256, 256, 0, stream>>>(W, wbf);

    ctlstm_scan<<<NBLK, 256, 0, stream>>>(types, dtime, wbf, tbl,
                                          h0, h1, syncc, (float*)d_out);
}
